// roadtransformer_42150809043349
// MI455X (gfx1250) — matrix operand order repaired and checked by reading, not compiled, not run on hardware
//
#include <hip/hip_runtime.h>
#include <hip/hip_bf16.h>
#include <stddef.h>

// ---------------- types ----------------
typedef __attribute__((ext_vector_type(16))) _Float16 v16h;
typedef __attribute__((ext_vector_type(8)))  _Float16 v8h;
typedef __attribute__((ext_vector_type(2)))  _Float16 h2;
typedef __attribute__((ext_vector_type(8)))  float    v8f;

static __device__ inline v8f wmma_f16(v16h a, v16h b, v8f c) {
  // D = A(16x32 f16) * B(32x16 f16) + C(16x16 f32)
  return __builtin_amdgcn_wmma_f32_16x16x32_f16(false, a, false, b, (short)0, c, false, false);
}
static __device__ inline v16h mk16(v8h lo, v8h hi) {
  return __builtin_shufflevector(lo, hi, 0,1,2,3,4,5,6,7,8,9,10,11,12,13,14,15);
}
static __device__ inline h2 pick2(v8h v, int c) {  // c is compile-time under unroll
  return __builtin_shufflevector(v, v, 0, 1); // placeholder overwritten by explicit cases below
}

// Problem constants
#define CC   128   // IN_PLANES
#define HH   64    // H
#define O2   256   // OUT_PLANES*2
#define NG   8     // GROUPS
#define GP16 16    // GP
#define RVW  160   // padded relvB row width (>= 144 needed, zero-filled past 126)
#define SW   96    // skewed-S row width (dd in [0,79))

// workspace byte offsets
#define WQ16_OFF   0u        // f16[256*128]   65536
#define RELQ_OFF   65536u    // f16[127*8] pad 2048
#define RELK_OFF   67584u    // f16[127*8] pad 2048
#define RELV_OFF   69632u    // f16[16*160]    5120
#define BIAS_OFF   74752u    // f32[256]       1024
#define SSC_OFF    75776u    // f32[24] pad    128
#define SSH_OFF    75904u    // f32[8]  pad    128
#define OSC_OFF    76032u    // f32[256]       1024
#define OSH_OFF    77056u    // f32[256]       1024

// ---------------- prep: fold BN, cast to f16, pack relative tables ----------------
__global__ __launch_bounds__(256) void axial_prep(
    const float* __restrict__ w_qkv, const float* __restrict__ relative,
    const float* __restrict__ bnq, const float* __restrict__ bns, const float* __restrict__ bno,
    _Float16* __restrict__ wq16, _Float16* __restrict__ relq8, _Float16* __restrict__ relk8,
    _Float16* __restrict__ relvB, float* __restrict__ bias,
    float* __restrict__ sscale, float* __restrict__ sshift,
    float* __restrict__ oscale, float* __restrict__ oshift) {
  __shared__ float s_sc[O2];
  const int t = threadIdx.x;
  { // qkv BN: scale folded into W rows, shift becomes bias
    float gm = bnq[t], bt = bnq[256 + t], mn = bnq[512 + t], vr = bnq[768 + t];
    float sc = gm * rsqrtf(vr + 1e-5f);
    s_sc[t] = sc;
    bias[t] = bt - sc * mn;
  }
  if (t < 24) { // sim BN scales, layout [qk(8)|qr(8)|kr(8)]
    float gm = bns[t], vr = bns[72 + t];
    sscale[t] = gm * rsqrtf(vr + 1e-5f);
  }
  if (t < 8) { // summed shifts of the 3 BN branches per group
    float sh = 0.f;
    for (int tt = 0; tt < 3; ++tt) {
      int ch = tt * 8 + t;
      float gm = bns[ch], bt = bns[24 + ch], mn = bns[48 + ch], vr = bns[72 + ch];
      float sc = gm * rsqrtf(vr + 1e-5f);
      sh += bt - sc * mn;
    }
    sshift[t] = sh;
  }
  { // output BN (256 channels)
    float gm = bno[t], bt = bno[256 + t], mn = bno[512 + t], vr = bno[768 + t];
    float sc = gm * rsqrtf(vr + 1e-5f);
    oscale[t] = sc;
    oshift[t] = bt - sc * mn;
  }
  __syncthreads();
  // W f16 with folded scale
  for (int kk = 0; kk < 128; ++kk) {
    int idx = kk * 256 + t;           // 256*128
    int o = idx >> 7;
    wq16[idx] = (_Float16)(w_qkv[idx] * s_sc[o]);
  }
  // packed q/k relative rows: relq8[d*8+c] = relative[c, d]
  for (int kk = 0; kk < 4; ++kk) {
    int idx = kk * 256 + t;
    if (idx < 127 * 8) {
      int d = idx >> 3, c = idx & 7;
      relq8[idx] = (_Float16)relative[c * 127 + d];
      relk8[idx] = (_Float16)relative[(8 + c) * 127 + d];
    }
  }
  // v-embedding as WMMA B operand: relvB[c*RVW+d] = relative[16+c, d], zero-padded d>=127
  for (int kk = 0; kk < 10; ++kk) {
    int idx = kk * 256 + t;           // 16*160 = 2560
    if (idx < 16 * RVW) {
      int c = idx / RVW, d = idx % RVW;
      relvB[idx] = (_Float16)((d < 127) ? relative[(16 + c) * 127 + d] : 0.f);
    }
  }
}

// ---------------- main fused kernel: one block per (n,w) ----------------
__global__ __launch_bounds__(256) void axial_main(
    const float* __restrict__ x,
    const _Float16* __restrict__ wq16, const _Float16* __restrict__ relq8,
    const _Float16* __restrict__ relk8, const _Float16* __restrict__ relvB,
    const float* __restrict__ bias,
    const float* __restrict__ sscale, const float* __restrict__ sshift,
    const float* __restrict__ oscale, const float* __restrict__ oshift,
    float* __restrict__ out) {
  // LDS: 72KB total
  __shared__ __align__(16) _Float16 s_xt[HH * CC];        // 16KB: xT[h][c]; per-wave sim[16][64] in phase 3
  __shared__ __align__(16) _Float16 s_qT[NG * HH * 8];    // 8KB : qT[g][i][c]
  __shared__ __align__(16) _Float16 s_kB[NG * HH * 8];    // 8KB : kB[g][j][c]
  __shared__ __align__(16) _Float16 s_v [NG * GP16 * HH]; // 16KB: v[g][c][j]
  __shared__ __align__(16) _Float16 s_S [NG * 16 * SW];   // 24KB: per-wave skewed sim (A of sve)

  const int tid  = threadIdx.x;
  const int lane = tid & 31;
  const int wv   = tid >> 5;            // wave id 0..7
  const bool lo16 = lane < 16;
  const int n15  = lane & 15;
  const int b    = blockIdx.x;          // b = n*64 + w
  const int nb   = b >> 6, wb = b & 63;

  // ---- phase 1: gather x[n,:,:,w] -> LDS f16, layout xT[h][c] ----
  const float* xbase = x + (size_t)nb * (CC * HH * 64) + wb;
  for (int kk = 0; kk < 32; ++kk) {
    int idx = kk * 256 + tid;           // 8192 = 128c * 64h
    int c = idx & 127, h = idx >> 7;
    s_xt[h * CC + c] = (_Float16)xbase[(size_t)c * (HH * 64) + h * 64];
  }
  __syncthreads();

  // ---- phase 2: qkv = Wf(256x128) @ x(128x64) + bias; wave w produces its own group's q/k/v ----
  // even tile (rows 32w..32w+15): lanes<16 -> q[c=r], lanes>=16 -> k[c=r]
  // odd  tile (rows 32w+16..+31): all lanes -> v[c = r + 8*half]
  for (int par = 0; par < 2; ++par) {
    const int mt = wv * 2 + par;
    for (int nt = 0; nt < 4; ++nt) {
      v8f acc = {};
      #pragma unroll
      for (int kt = 0; kt < 4; ++kt) {
        int kbA = kt * 32 + (lo16 ? 0 : 8);
        const _Float16* ap = wq16 + (mt * 16 + n15) * CC + kbA;
        v16h a = mk16(*(const v8h*)ap, *(const v8h*)(ap + 16));
        int kbB = kt * 32 + (lo16 ? 0 : 8);
        const _Float16* bp = s_xt + (nt * 16 + n15) * CC + kbB;
        v16h bb = mk16(*(const v8h*)bp, *(const v8h*)(bp + 16));
        acc = wmma_f16(a, bb, acc);
      }
      const int h = nt * 16 + n15;
      if (par == 0) {
        _Float16* pb = (lo16 ? s_qT : s_kB) + (wv * 64 + h) * 8;
        #pragma unroll
        for (int r = 0; r < 8; ++r) {
          int o = wv * 32 + r + (lo16 ? 0 : 8);
          pb[r] = (_Float16)(acc[r] + bias[o]);
        }
      } else {
        _Float16* pv = s_v + wv * 16 * 64 + (lo16 ? 0 : 8 * 64) + h;
        #pragma unroll
        for (int r = 0; r < 8; ++r) {
          int o = wv * 32 + 16 + r + (lo16 ? 0 : 8);
          pv[r * 64] = (_Float16)(acc[r] + bias[o]);
        }
      }
    }
  }
  __syncthreads();

  // ---- phase 3: wave g handles group g ----
  const int g = wv;
  _Float16* qTg  = s_qT + g * 512;
  _Float16* kBg  = s_kB + g * 512;
  _Float16* vg   = s_v  + g * 1024;
  _Float16* simw = s_xt + wv * 1024;     // 16x64 f16 per wave (xT dead)
  _Float16* Sw   = s_S  + wv * 16 * SW;  // 16x96 f16 skew buffer
  const float sqk = sscale[g], sqr = sscale[8 + g], skr = sscale[16 + g], ssh = sshift[g];

  for (int mt = 0; mt < 4; ++mt) {
    // qk = q^T k  (K padded 8->32 with zeros)
    v16h aq = {};
    if (lo16) aq = mk16(*(const v8h*)(qTg + (mt * 16 + n15) * 8), (v8h){});
    v8f st[4];
    v8h krow[4];
    #pragma unroll
    for (int nt = 0; nt < 4; ++nt) {
      krow[nt] = *(const v8h*)(kBg + (nt * 16 + n15) * 8);
      v16h bk = {};
      if (lo16) {
        #pragma unroll
        for (int i = 0; i < 8; ++i) bk[i] = krow[nt][i];
      }
      v8f z = {};
      st[nt] = wmma_f16(aq, bk, z);
    }
    // + qr/kr (Toeplitz terms, packed f16 fma) + BN scales/shift
    #pragma unroll
    for (int r = 0; r < 8; ++r) {
      const int i = mt * 16 + r + (lo16 ? 0 : 8);
      const v8h qrow = *(const v8h*)(qTg + i * 8);
      #pragma unroll
      for (int nt = 0; nt < 4; ++nt) {
        const int j = nt * 16 + n15;
        const h2* rq2 = (const h2*)(relq8 + (i - j + 63) * 8);
        const h2* rk2 = (const h2*)(relk8 + (j - i + 63) * 8);
        h2 aq2 = {}, ak2 = {};
        #pragma unroll
        for (int c = 0; c < 4; ++c) {
          h2 qc = __builtin_shufflevector(qrow, qrow, 0, 1);
          h2 kc = __builtin_shufflevector(krow[nt], krow[nt], 0, 1);
          if (c == 1) { qc = __builtin_shufflevector(qrow, qrow, 2, 3); kc = __builtin_shufflevector(krow[nt], krow[nt], 2, 3); }
          if (c == 2) { qc = __builtin_shufflevector(qrow, qrow, 4, 5); kc = __builtin_shufflevector(krow[nt], krow[nt], 4, 5); }
          if (c == 3) { qc = __builtin_shufflevector(qrow, qrow, 6, 7); kc = __builtin_shufflevector(krow[nt], krow[nt], 6, 7); }
          aq2 = qc * rq2[c] + aq2;   // v_pk_fma_f16
          ak2 = kc * rk2[c] + ak2;
        }
        float qr = (float)aq2[0] + (float)aq2[1];
        float kr = (float)ak2[0] + (float)ak2[1];
        st[nt][r] = fmaf(sqk, st[nt][r], fmaf(sqr, qr, fmaf(skr, kr, ssh)));
      }
    }
    // row softmax (row spread over 16 lanes x 4 nt)
    #pragma unroll
    for (int r = 0; r < 8; ++r) {
      float mx = fmaxf(fmaxf(st[0][r], st[1][r]), fmaxf(st[2][r], st[3][r]));
      for (int msk = 1; msk <= 8; msk <<= 1) mx = fmaxf(mx, __shfl_xor(mx, msk, 32));
      float sum = 0.f;
      #pragma unroll
      for (int nt = 0; nt < 4; ++nt) { float e = __expf(st[nt][r] - mx); st[nt][r] = e; sum += e; }
      for (int msk = 1; msk <= 8; msk <<= 1) sum += __shfl_xor(sum, msk, 32);
      float inv = 1.f / sum;
      #pragma unroll
      for (int nt = 0; nt < 4; ++nt) st[nt][r] *= inv;
    }
    // zero skew buffer (16*96 per wave, in-order LDS makes this safe pre-scatter)
    #pragma unroll
    for (int z = 0; z < 6; ++z) { v8h zz = {}; *(v8h*)(Sw + (z * 32 + lane) * 8) = zz; }
    // write sim rowblock (A of sv) and skewed S (A of sve), both unconditional
    #pragma unroll
    for (int r = 0; r < 8; ++r) {
      const int iloc = r + (lo16 ? 0 : 8);
      #pragma unroll
      for (int nt = 0; nt < 4; ++nt) {
        const int j = nt * 16 + n15;
        _Float16 hv = (_Float16)st[nt][r];
        simw[iloc * 64 + j] = hv;
        Sw[iloc * SW + (iloc + 63 - j)] = hv;   // dd = i - j + 63 - mt*16 in [0,79)
      }
    }
    // sv = sim @ v^T  (16x16, K=64)
    v8f sv = {};
    #pragma unroll
    for (int ks = 0; ks < 2; ++ks) {
      int ka = ks * 32 + (lo16 ? 0 : 8);
      const _Float16* ap = simw + n15 * 64 + ka;
      v16h a = mk16(*(const v8h*)ap, *(const v8h*)(ap + 16));
      int kb = ks * 32 + (lo16 ? 0 : 8);
      const _Float16* bp = vg + n15 * 64 + kb;
      v16h bb = mk16(*(const v8h*)bp, *(const v8h*)(bp + 16));
      sv = wmma_f16(a, bb, sv);
    }
    // sve^T = S(16x96) @ relvB^T slice (K=96, rows offset by mt*16; padded zeros past d=126)
    v8f sve = {};
    #pragma unroll
    for (int ks = 0; ks < 3; ++ks) {
      int ka = ks * 32 + (lo16 ? 0 : 8);
      const _Float16* ap = Sw + n15 * SW + ka;
      v16h a = mk16(*(const v8h*)ap, *(const v8h*)(ap + 16));
      int kb = mt * 16 + ks * 32 + (lo16 ? 0 : 8);
      const _Float16* bp = relvB + n15 * RVW + kb;
      v16h bb = mk16(*(const v8h*)bp, *(const v8h*)(bp + 16));
      sve = wmma_f16(a, bb, sve);
    }
    // output BN pair-sum + store: out[n, p=g*16+c, i, w]
    const int p = g * 16 + n15;
    const float sc0 = oscale[2 * p], sh0 = oshift[2 * p];
    const float sc1 = oscale[2 * p + 1], sh1 = oshift[2 * p + 1];
    #pragma unroll
    for (int r = 0; r < 8; ++r) {
      const int iG = mt * 16 + r + (lo16 ? 0 : 8);
      float oA = fmaf(sc0, sv[r], sh0) + fmaf(sc1, sve[r], sh1);
      out[(((size_t)nb * 128 + p) * 64 + iG) * 64 + wb] = oA;
    }
  }
}

extern "C" void kernel_launch(void* const* d_in, const int* in_sizes, int n_in,
                              void* d_out, int out_size, void* d_ws, size_t ws_size,
                              hipStream_t stream) {
  (void)in_sizes; (void)n_in; (void)out_size; (void)ws_size;
  const float* x        = (const float*)d_in[0];
  const float* w_qkv    = (const float*)d_in[1];
  const float* relative = (const float*)d_in[2];
  const float* bnq      = (const float*)d_in[3];
  const float* bns      = (const float*)d_in[4];
  const float* bno      = (const float*)d_in[5];
  float* out = (float*)d_out;
  char* ws = (char*)d_ws;

  _Float16* wq16  = (_Float16*)(ws + WQ16_OFF);
  _Float16* relq8 = (_Float16*)(ws + RELQ_OFF);
  _Float16* relk8 = (_Float16*)(ws + RELK_OFF);
  _Float16* relvB = (_Float16*)(ws + RELV_OFF);
  float* bias   = (float*)(ws + BIAS_OFF);
  float* sscale = (float*)(ws + SSC_OFF);
  float* sshift = (float*)(ws + SSH_OFF);
  float* oscale = (float*)(ws + OSC_OFF);
  float* oshift = (float*)(ws + OSH_OFF);

  axial_prep<<<1, 256, 0, stream>>>(w_qkv, relative, bnq, bns, bno,
                                    wq16, relq8, relk8, relvB,
                                    bias, sscale, sshift, oscale, oshift);
  axial_main<<<1024, 256, 0, stream>>>(x, wq16, relq8, relk8, relvB,
                                       bias, sscale, sshift, oscale, oshift, out);
}
